// feat_extraction_layer_66623532695715
// MI455X (gfx1250) — compile-verified, matrix operand order checked by reading, not run on hardware
//
#include <hip/hip_runtime.h>
#include <hip/hip_bf16.h>

// ---------------------------------------------------------------------------
// PointNet++ set-abstraction layer for MI455X (gfx1250, wave32, WMMA)
// B=16, N=16384, NPOINT=512, NSAMPLE=16, radius=0.1
// ---------------------------------------------------------------------------

#define BB      16
#define NN      16384
#define NPOINT  512
#define NSAMPLE 16
#define RAD2    0.01f
#define NROWS   (BB * NPOINT * NSAMPLE)   // 131072 feature rows
#define CHUNK   2048                      // ball-query LDS staging chunk

typedef __attribute__((ext_vector_type(16))) _Float16 v16h;
typedef __attribute__((ext_vector_type(8)))  float    v8f;

// ---------------------------------------------------------------------------
// CDNA5 async global->LDS copy helpers (guarded; fall back to sync copy).
// Probe: builtin exists with signature (AS1 int*, AS3 int*, i32, i32).
// ---------------------------------------------------------------------------
#if defined(__has_builtin)
#if __has_builtin(__builtin_amdgcn_global_load_async_to_lds_b32)
#define HAVE_ASYNC_LDS 1
#endif
#endif

typedef __attribute__((address_space(1))) int* gas_i32p;
typedef __attribute__((address_space(3))) int* las_i32p;

__device__ __forceinline__ void stage_f32(const float* __restrict__ g, float* l)
{
#ifdef HAVE_ASYNC_LDS
    __builtin_amdgcn_global_load_async_to_lds_b32(
        (gas_i32p)g, (las_i32p)l, 0, 0);
#else
    *l = *g;
#endif
}

__device__ __forceinline__ void stage_wait()
{
#ifdef HAVE_ASYNC_LDS
#if defined(__has_builtin) && __has_builtin(__builtin_amdgcn_s_wait_asynccnt)
    __builtin_amdgcn_s_wait_asynccnt(0);
#else
    asm volatile("s_wait_asynccnt 0x0" ::: "memory");
#endif
#endif
}

// ---------------------------------------------------------------------------
// Kernel 1: farthest point sampling. One block (1024 thr) per batch.
// Coordinates + running min-distance live in registers (16 pts/thread).
// Argmax: wave shfl_xor reduce, then 32-wide cross-wave reduce.
// Also emits new_xyz (both workspace row-major and transposed output).
// ---------------------------------------------------------------------------
__global__ __launch_bounds__(1024) void fps_kernel(
    const float* __restrict__ pts, int* __restrict__ fpsIdx,
    float* __restrict__ newXyz, float* __restrict__ outXyz)
{
    const int b    = blockIdx.x;
    const int tid  = threadIdx.x;
    const int lane = tid & 31;
    const int wv   = tid >> 5;

    const float* px = pts + (size_t)b * 6 * NN;
    const float* py = px + NN;
    const float* pz = px + 2 * NN;

    float x[16], y[16], z[16], dist[16];
    #pragma unroll
    for (int k = 0; k < 16; ++k) {
        int n = tid + k * 1024;
        x[k] = px[n]; y[k] = py[n]; z[k] = pz[n];
        dist[k] = 1e10f;
    }

    __shared__ float cxyz[3];
    __shared__ float wrv[32];
    __shared__ int   wri[32];
    __shared__ int   sfar;

    int farthest = 0;
    for (int i = 0; i < NPOINT; ++i) {
        if (tid == 0) {
            fpsIdx[b * NPOINT + i] = farthest;
            float cx = px[farthest], cy = py[farthest], cz = pz[farthest];
            cxyz[0] = cx; cxyz[1] = cy; cxyz[2] = cz;
            float* nx = newXyz + (size_t)(b * NPOINT + i) * 3;
            nx[0] = cx; nx[1] = cy; nx[2] = cz;
            // output new_xyz transposed (B, 3, NPOINT)
            outXyz[b * 3 * NPOINT + 0 * NPOINT + i] = cx;
            outXyz[b * 3 * NPOINT + 1 * NPOINT + i] = cy;
            outXyz[b * 3 * NPOINT + 2 * NPOINT + i] = cz;
        }
        __syncthreads();
        float cx = cxyz[0], cy = cxyz[1], cz = cxyz[2];

        float bv = -1.0f; int bi = 0;
        #pragma unroll
        for (int k = 0; k < 16; ++k) {
            float dx = x[k] - cx, dy = y[k] - cy, dz = z[k] - cz;
            float d  = dx * dx + dy * dy + dz * dz;
            float nd = fminf(dist[k], d);
            dist[k] = nd;
            int n = tid + k * 1024;
            if (nd > bv || (nd == bv && n < bi)) { bv = nd; bi = n; }
        }
        // wave32 argmax reduce (first-index tie break to match jnp.argmax)
        #pragma unroll
        for (int off = 16; off > 0; off >>= 1) {
            float ov = __shfl_xor(bv, off, 32);
            int   oi = __shfl_xor(bi, off, 32);
            if (ov > bv || (ov == bv && oi < bi)) { bv = ov; bi = oi; }
        }
        if (lane == 0) { wrv[wv] = bv; wri[wv] = bi; }
        __syncthreads();
        if (wv == 0) {
            bv = wrv[lane]; bi = wri[lane];
            #pragma unroll
            for (int off = 16; off > 0; off >>= 1) {
                float ov = __shfl_xor(bv, off, 32);
                int   oi = __shfl_xor(bi, off, 32);
                if (ov > bv || (ov == bv && oi < bi)) { bv = ov; bi = oi; }
            }
            if (lane == 0) sfar = bi;
        }
        __syncthreads();
        farthest = sfar;
    }
}

// ---------------------------------------------------------------------------
// Kernel 2: ball query + grouping. One wave per (b, s) centroid.
// All 8 waves of a block share one batch, so point chunks are staged into
// LDS once per block via CDNA5 async global->LDS loads (ASYNCcnt), then each
// wave tests its centroid against the LDS copy. Ballot prefix-sum collects
// the first NSAMPLE in-radius indices in index order (== reference's
// sort-then-truncate), tail filled with idx[0].
// ---------------------------------------------------------------------------
__global__ __launch_bounds__(256) void ballquery_kernel(
    const float* __restrict__ pts, const float* __restrict__ newXyz,
    float* __restrict__ feat)
{
    const int lane = threadIdx.x & 31;
    const int warp = threadIdx.x >> 5;
    const int wid  = blockIdx.x * 8 + warp;      // 0..8191 == b*512+s
    const int b    = wid >> 9;                   // same for all 8 waves

    __shared__ float sx[CHUNK], sy[CHUNK], sz[CHUNK];
    __shared__ int   lists[8][NSAMPLE];
    __shared__ int   sdone;

    const float* px = pts + (size_t)b * 6 * NN;
    const float* py = px + NN;
    const float* pz = px + 2 * NN;

    const float qx = newXyz[wid * 3 + 0];
    const float qy = newXyz[wid * 3 + 1];
    const float qz = newXyz[wid * 3 + 2];

    if (threadIdx.x == 0) sdone = 0;

    int  count = 0;
    bool done  = false;
    for (int base = 0; base < NN; base += CHUNK) {
        __syncthreads();   // previous chunk fully consumed; sdone visible
        // cooperative stage of CHUNK points (x,y,z) into LDS
        for (int i = threadIdx.x; i < CHUNK; i += 256) {
            stage_f32(px + base + i, &sx[i]);
            stage_f32(py + base + i, &sy[i]);
            stage_f32(pz + base + i, &sz[i]);
        }
        stage_wait();
        __syncthreads();

        if (!done) {
            for (int off = 0; off < CHUNK && count < NSAMPLE; off += 32) {
                int j = off + lane;
                float dx = sx[j] - qx, dy = sy[j] - qy, dz = sz[j] - qz;
                bool inb = (dx * dx + dy * dy + dz * dz) <= RAD2;
                unsigned mask = (unsigned)__ballot(inb);
                if (inb) {
                    int pos = count + __popc(mask & ((1u << lane) - 1u));
                    if (pos < NSAMPLE) lists[warp][pos] = base + j;
                }
                count += __popc(mask);
            }
            if (count >= NSAMPLE) {
                done = true;
                if (lane == 0) atomicAdd(&sdone, 1);
            }
        }
        __syncthreads();
        if (sdone == 8) break;   // uniform: all waves filled
    }
    __builtin_amdgcn_wave_barrier();
    __threadfence_block();
    if (count < 1) { if (lane == 0) lists[warp][0] = 0; count = 1; }
    __threadfence_block();

    if (lane < NSAMPLE) {
        int pj = (lane < count) ? lists[warp][lane] : lists[warp][0];
        float gx = px[pj] - qx, gy = py[pj] - qy, gz = pz[pj] - qz;
        float nx = px[3 * NN + pj], ny = px[4 * NN + pj], nz = px[5 * NN + pj];
        float* f = feat + ((size_t)wid * NSAMPLE + lane) * 6;
        f[0] = gx; f[1] = gy; f[2] = gz; f[3] = nx; f[4] = ny; f[5] = nz;
    }
}

// ---------------------------------------------------------------------------
// Kernel 3 (x3 instantiations): pointwise MLP layer using
// V_WMMA_F32_16X16X32_F16. One wave computes a 16-row x NC-col tile.
// Optional fused BN(scale,shift)+ReLU on the input, fused per-channel
// sum / sum-of-squares accumulation for the *output* BN stats.
// Conv bias skipped: it cancels inside batch-norm.
// ---------------------------------------------------------------------------
template <int KD, int NC, bool BN>
__global__ __launch_bounds__(256) void mlp_wmma_kernel(
    const float* __restrict__ in, const float* __restrict__ sc,
    const float* __restrict__ sh, const float* __restrict__ w,
    float* __restrict__ out, float* __restrict__ stats)
{
    constexpr int NT = NC / 16;
    const int lane   = threadIdx.x & 31;
    const int m      = lane & 15;
    const int hi     = lane >> 4;
    const int wave   = blockIdx.x * (blockDim.x >> 5) + (threadIdx.x >> 5);
    const int nwaves = gridDim.x * (blockDim.x >> 5);

    // per-lane K slots of the 16-bit A/B fragment layout (ISA 7.12.2)
    int kb[8];
    #pragma unroll
    for (int v = 0; v < 8; ++v)
        kb[v] = (v < 4) ? ((hi ? 8 : 0) + 2 * v) : ((hi ? 24 : 16) + 2 * (v - 4));

    float asc[16], ash[16];
    if (BN) {
        #pragma unroll
        for (int v = 0; v < 8; ++v) {
            int k0 = kb[v];
            asc[2 * v]     = (k0 < KD)     ? sc[k0]     : 0.0f;
            ash[2 * v]     = (k0 < KD)     ? sh[k0]     : 0.0f;
            asc[2 * v + 1] = (k0 + 1 < KD) ? sc[k0 + 1] : 0.0f;
            ash[2 * v + 1] = (k0 + 1 < KD) ? sh[k0 + 1] : 0.0f;
        }
    }

    // B fragments: column n = t*16 + m per lane; B[k][n] = w[n*KD + k]
    v16h breg[NT];
    #pragma unroll
    for (int t = 0; t < NT; ++t) {
        v16h bb = {};
        int ncol = t * 16 + m;
        #pragma unroll
        for (int v = 0; v < 8; ++v) {
            int k0 = kb[v];
            bb[2 * v]     = (k0 < KD)     ? (_Float16)w[ncol * KD + k0]     : (_Float16)0.0f;
            bb[2 * v + 1] = (k0 + 1 < KD) ? (_Float16)w[ncol * KD + k0 + 1] : (_Float16)0.0f;
        }
        breg[t] = bb;
    }

    float sum[NT], sq[NT];
    #pragma unroll
    for (int t = 0; t < NT; ++t) { sum[t] = 0.0f; sq[t] = 0.0f; }

    const int NTILES = NROWS / 16;   // 8192; 2048 waves -> 4 tiles each (uniform)
    for (int tile = wave; tile < NTILES; tile += nwaves) {
        const float* rowp = in + (size_t)(tile * 16 + m) * KD;
        // prefetch next tile's rows (global_prefetch_b8); speculative-safe
        __builtin_prefetch(rowp + (size_t)nwaves * 16 * KD, 0, 1);

        v16h a;
        #pragma unroll
        for (int v = 0; v < 8; ++v) {
            int k0 = kb[v];
            float x0 = (k0 < KD)     ? rowp[k0]     : 0.0f;
            float x1 = (k0 + 1 < KD) ? rowp[k0 + 1] : 0.0f;
            if (BN) {
                x0 = fmaxf(x0 * asc[2 * v]     + ash[2 * v],     0.0f);
                x1 = fmaxf(x1 * asc[2 * v + 1] + ash[2 * v + 1], 0.0f);
            }
            a[2 * v]     = (_Float16)x0;
            a[2 * v + 1] = (_Float16)x1;
        }
        #pragma unroll
        for (int t = 0; t < NT; ++t) {
            v8f c = {};
            c = __builtin_amdgcn_wmma_f32_16x16x32_f16(
                    false, a, false, breg[t], (short)0, c, false, false);
            int ncol = t * 16 + m;
            float* op = out + (size_t)tile * 16 * NC;
            #pragma unroll
            for (int r = 0; r < 8; ++r) {
                float d   = c[r];
                int mrow  = hi ? (r + 8) : r;   // D layout: lanes 16-31 hold M+8
                op[(size_t)mrow * NC + ncol] = d;
                sum[t] += d;
                sq[t]  += d * d;
            }
        }
    }
    #pragma unroll
    for (int t = 0; t < NT; ++t) {
        int ncol = t * 16 + m;
        atomicAdd(&stats[ncol],      sum[t]);
        atomicAdd(&stats[NC + ncol], sq[t]);
    }
}

// ---------------------------------------------------------------------------
// Kernel 4: fold raw (sum, sumsq) + gamma/beta into BN scale & shift.
// ---------------------------------------------------------------------------
__global__ void finalize_stats(const float* __restrict__ raw,
                               const float* __restrict__ g,
                               const float* __restrict__ be,
                               float* __restrict__ sc, float* __restrict__ sh,
                               int nc)
{
    int n = threadIdx.x;
    if (n >= nc) return;
    const float R = (float)NROWS;
    float mean = raw[n] / R;
    float var  = raw[nc + n] / R - mean * mean;   // biased var, matches jnp.var
    float inv  = rsqrtf(var + 1e-5f);
    float s    = g[n] * inv;
    sc[n] = s;
    sh[n] = be[n] - mean * s;
}

// ---------------------------------------------------------------------------
// Kernel 5: BN3 + ReLU + max over nsample, write transposed (B, 32, NPOINT).
// ---------------------------------------------------------------------------
__global__ __launch_bounds__(256) void maxpool_kernel(
    const float* __restrict__ h3, const float* __restrict__ sc,
    const float* __restrict__ sh, float* __restrict__ outNP)
{
    int idx = blockIdx.x * blockDim.x + threadIdx.x;
    if (idx >= BB * NPOINT * 32) return;
    int o = idx & 31;
    int s = (idx >> 5) & (NPOINT - 1);
    int b = idx >> 14;
    float scale = sc[o], shift = sh[o];
    const float* base = h3 + ((size_t)(b * NPOINT + s) * NSAMPLE) * 32 + o;
    float mx = 0.0f;   // all ReLU outputs >= 0
    #pragma unroll
    for (int j = 0; j < NSAMPLE; ++j) {
        float v = fmaxf(base[(size_t)j * 32] * scale + shift, 0.0f);
        mx = fmaxf(mx, v);
    }
    outNP[(size_t)b * 32 * NPOINT + o * NPOINT + s] = mx;
}

// ---------------------------------------------------------------------------
extern "C" void kernel_launch(void* const* d_in, const int* in_sizes, int n_in,
                              void* d_out, int out_size, void* d_ws, size_t ws_size,
                              hipStream_t stream)
{
    (void)in_sizes; (void)n_in; (void)out_size; (void)ws_size;

    const float* pts = (const float*)d_in[0];
    const float* w1  = (const float*)d_in[1];
    const float* g1  = (const float*)d_in[3];
    const float* be1 = (const float*)d_in[4];
    const float* w2  = (const float*)d_in[5];
    const float* g2  = (const float*)d_in[7];
    const float* be2 = (const float*)d_in[8];
    const float* w3  = (const float*)d_in[9];
    const float* g3  = (const float*)d_in[11];
    const float* be3 = (const float*)d_in[12];
    float* out = (float*)d_out;

    char*  ws     = (char*)d_ws;
    float* stats  = (float*)ws;                      // 256 floats (raw + params)
    int*   fpsIdx = (int*)(ws + 1024);               // 8192 ints
    float* newXyz = (float*)(ws + 1024 + 32768);     // 24576 floats (b,s,3)
    float* feat   = newXyz + (size_t)BB * NPOINT * 3;       // NROWS x 6
    float* h1     = feat   + (size_t)NROWS * 6;             // NROWS x 16
    float* h2     = h1     + (size_t)NROWS * 16;            // NROWS x 16
    float* h3     = h2     + (size_t)NROWS * 16;            // NROWS x 32

    float* s1  = stats;       float* s2  = stats + 32;  float* s3  = stats + 64;
    float* sc1 = stats + 128; float* sh1 = stats + 144;
    float* sc2 = stats + 160; float* sh2 = stats + 176;
    float* sc3 = stats + 192; float* sh3 = stats + 224;

    (void)hipMemsetAsync(stats, 0, 1024, stream);

    fps_kernel<<<BB, 1024, 0, stream>>>(pts, fpsIdx, newXyz, out);
    ballquery_kernel<<<(BB * NPOINT) / 8, 256, 0, stream>>>(pts, newXyz, feat);

    mlp_wmma_kernel<6, 16, false><<<256, 256, 0, stream>>>(feat, nullptr, nullptr, w1, h1, s1);
    finalize_stats<<<1, 32, 0, stream>>>(s1, g1, be1, sc1, sh1, 16);

    mlp_wmma_kernel<16, 16, true><<<256, 256, 0, stream>>>(h1, sc1, sh1, w2, h2, s2);
    finalize_stats<<<1, 32, 0, stream>>>(s2, g2, be2, sc2, sh2, 16);

    mlp_wmma_kernel<16, 32, true><<<256, 256, 0, stream>>>(h2, sc2, sh2, w3, h3, s3);
    finalize_stats<<<1, 32, 0, stream>>>(s3, g3, be3, sc3, sh3, 32);

    maxpool_kernel<<<(BB * NPOINT * 32 + 255) / 256, 256, 0, stream>>>(
        h3, sc3, sh3, out + BB * 3 * NPOINT);
}
